// LinGCN_64424509440205
// MI455X (gfx1250) — matrix operand order repaired and checked by reading, not compiled, not run on hardware
//
#include <hip/hip_runtime.h>
#include <hip/hip_bf16.h>
#include <math.h>

typedef __attribute__((ext_vector_type(16))) _Float16 v16h;
typedef __attribute__((ext_vector_type(4)))  _Float16 v4h;
typedef __attribute__((ext_vector_type(2)))  _Float16 v2h;
typedef __attribute__((ext_vector_type(8)))  float    v8f;

// ---------------------------------------------------------------------------
// Degree / normalization kernels
// ---------------------------------------------------------------------------
__global__ void k_fill_ones(float* p, int n) {
    int i = blockIdx.x * blockDim.x + threadIdx.x;
    if (i < n) p[i] = 1.0f;                      // self-loop contributes 1 to deg
}

__global__ void k_deg_accum(const int* __restrict__ dst, int E, float* __restrict__ deg) {
    int i = blockIdx.x * blockDim.x + threadIdx.x;
    if (i < E) atomicAdd(&deg[dst[i]], 1.0f);
}

__global__ void k_rsqrt(float* p, int n) {
    int i = blockIdx.x * blockDim.x + threadIdx.x;
    if (i < n) p[i] = rsqrtf(p[i]);              // deg >= 1 always (self loops)
}

// ---------------------------------------------------------------------------
// Tiled WMMA GEMM:  C[Nrows x Mreal] = op(A[Nrows x K]) * W[K x Mreal] (+bias)
//   flags bit0: relu epilogue        flags bit1: relu applied to A on load
// Block: 64 output rows x MPAD output cols; 32*(MPAD/16) threads; one wave per
// 16-col tile, 4 row tiles per wave.  K staged to LDS in f16 chunks of 64,
// ping-pong double buffered (one barrier per chunk, staging overlaps compute).
// W is staged TRANSPOSED (sWt[col][k]) so each lane's B fragment is two
// contiguous 16B runs -> ds_load_b128, matching the A-fragment path.
// ---------------------------------------------------------------------------
template <int MPAD>
__global__ __launch_bounds__(32 * (MPAD / 16))
void k_gemm_wmma(const float* __restrict__ A, int lda,
                 const float* __restrict__ W, int K, int Mreal,
                 const float* __restrict__ bias,
                 float* __restrict__ C, int ldc, int col_off,
                 int Nrows, int flags)
{
    constexpr int NT = MPAD / 16;
    constexpr int KC = 64;
    __shared__ _Float16 sA [2][64]  [KC + 8];   // activations, row-major
    __shared__ _Float16 sWt[2][MPAD][KC + 8];   // weights, transposed

    const int tid  = threadIdx.x;
    const int nthr = 32 * NT;
    const int wave = tid >> 5;
    const int lane = tid & 31;
    const int rowBase = blockIdx.x * 64;

    const int halfid = lane >> 4;        // 0: lanes 0-15, 1: lanes 16-31
    const int rr     = lane & 15;
    const int myCol  = wave * 16 + rr;   // B-fragment column for this lane

    // ---- staging helpers (packed LDS stores) ------------------------------
    auto stageW = [&](int kc, int buf) {
        // two consecutive K rows per thread at a fixed column -> one b32 store
        for (int idx = tid; idx < (KC / 2) * MPAD; idx += nthr) {
            int c  = idx % MPAD;
            int kr = (idx / MPAD) * 2;
            float v0 = 0.0f, v1 = 0.0f;
            if (c < Mreal) {
                v0 = W[(size_t)(kc + kr)     * Mreal + c];
                v1 = W[(size_t)(kc + kr + 1) * Mreal + c];
            }
            v2h pk; pk[0] = (_Float16)v0; pk[1] = (_Float16)v1;
            *(v2h*)&sWt[buf][c][kr] = pk;
        }
    };
    auto stageA = [&](int kc, int buf) {
        // float4 global load -> packed 4xf16 (8B) LDS store
        for (int idx = tid; idx < 64 * (KC / 4); idx += nthr) {
            int r  = idx / (KC / 4);
            int k4 = (idx % (KC / 4)) * 4;
            int gr = rowBase + r;
            float4 v = make_float4(0.0f, 0.0f, 0.0f, 0.0f);
            if (gr < Nrows) {
                v = *(const float4*)(A + (size_t)gr * lda + kc + k4);
                if (kc + KC < K)   // prefetch chunk-after-next (global_prefetch_b8)
                    __builtin_prefetch(A + (size_t)gr * lda + kc + KC + k4, 0, 0);
            }
            if (flags & 2) {
                v.x = fmaxf(v.x, 0.0f); v.y = fmaxf(v.y, 0.0f);
                v.z = fmaxf(v.z, 0.0f); v.w = fmaxf(v.w, 0.0f);
            }
            v4h pk;
            pk[0] = (_Float16)v.x; pk[1] = (_Float16)v.y;
            pk[2] = (_Float16)v.z; pk[3] = (_Float16)v.w;
            *(v4h*)&sA[buf][r][k4] = pk;
        }
    };

    v8f acc[4];
    #pragma unroll
    for (int rt = 0; rt < 4; ++rt) acc[rt] = v8f{};

    const int nc = K / KC;               // K is always a multiple of 64 here
    stageW(0, 0);
    stageA(0, 0);

    for (int c = 0; c < nc; ++c) {
        const int buf = c & 1;
        __syncthreads();                 // staging of buf done; safe to read

        #pragma unroll
        for (int kk = 0; kk < KC; kk += 32) {
            // B fragment: lane holds col myCol, K = kk + halfid*8 + {0..7}, kk + 16 + halfid*8 + {0..7}
            // -> two 16B runs = two ds_load_b128
            v16h b;
            #pragma unroll
            for (int e = 0; e < 16; ++e)
                b[e] = sWt[buf][myCol][kk + (e >> 3) * 16 + halfid * 8 + (e & 7)];

            #pragma unroll
            for (int rt = 0; rt < 4; ++rt) {
                // A fragment: row rr of tile rt; two contiguous 16B runs
                v16h a;
                #pragma unroll
                for (int e = 0; e < 8; ++e) {
                    a[e]     = sA[buf][rt * 16 + rr][kk + halfid * 8 + e];
                    a[e + 8] = sA[buf][rt * 16 + rr][kk + 16 + halfid * 8 + e];
                }
                acc[rt] = __builtin_amdgcn_wmma_f32_16x16x32_f16(
                    false, a, false, b, (short)0, acc[rt], false, false);
            }
        }

        if (c + 1 < nc) {                // stage next chunk into other buffer
            stageW((c + 1) * KC, buf ^ 1);
            stageA((c + 1) * KC, buf ^ 1);
        }
    }

    // epilogue: C/D layout VGPR i, lane l -> row = i + (l/16)*8, col = l%16
    if (myCol < Mreal) {
        float bv = bias ? bias[myCol] : 0.0f;
        #pragma unroll
        for (int rt = 0; rt < 4; ++rt) {
            #pragma unroll
            for (int i = 0; i < 8; ++i) {
                int grow = rowBase + rt * 16 + i + halfid * 8;
                if (grow < Nrows) {
                    float v = acc[rt][i] + bv;
                    if (flags & 1) v = v > 0.0f ? v : 0.0f;
                    C[(size_t)grow * ldc + col_off + myCol] = v;
                }
            }
        }
    }
}

// ---------------------------------------------------------------------------
// Self-loop init:  out[i, f] = dinv[i]^2 * H[i, f] (+ bias[f])      F = 128
// ---------------------------------------------------------------------------
__global__ void k_init_self(const float* __restrict__ dinv,
                            const float* __restrict__ H,
                            float* __restrict__ out, int ldo,
                            const float* __restrict__ bias, int Nn)
{
    int t = blockIdx.x * blockDim.x + threadIdx.x;
    int i = t >> 7, f = t & 127;
    if (i < Nn) {
        float dd = dinv[i];
        float v = dd * dd * H[(size_t)i * 128 + f];
        if (bias) v += bias[f];
        out[(size_t)i * ldo + f] = v;
    }
}

// ---------------------------------------------------------------------------
// Edge scatter: wave per edge, lane per 4 features (float4 gather, f32 atomics)
// ---------------------------------------------------------------------------
__global__ void k_scatter_edges(const int* __restrict__ src, const int* __restrict__ dst,
                                int E, const float* __restrict__ dinv,
                                const float* __restrict__ H,   // [N,128]
                                float* __restrict__ out, int ldo)
{
    int t = blockIdx.x * blockDim.x + threadIdx.x;
    int e = t >> 5, lane = t & 31;
    if (e >= E) return;
    int s = src[e], d = dst[e];
    float w = dinv[s] * dinv[d];
    float4 hv = *(const float4*)(H + (size_t)s * 128 + lane * 4);
    float* o = out + (size_t)d * ldo + lane * 4;
    atomicAdd(o + 0, w * hv.x);
    atomicAdd(o + 1, w * hv.y);
    atomicAdd(o + 2, w * hv.z);
    atomicAdd(o + 3, w * hv.w);
}

// ---------------------------------------------------------------------------
// bias + relu in place on an [N,128] buffer
// ---------------------------------------------------------------------------
__global__ void k_bias_relu(float* __restrict__ p, const float* __restrict__ bias, int Nn)
{
    int t = blockIdx.x * blockDim.x + threadIdx.x;
    int i = t >> 7, f = t & 127;
    if (i < Nn) {
        size_t o = (size_t)i * 128 + f;
        float v = p[o] + bias[f];
        p[o] = v > 0.0f ? v : 0.0f;
    }
}

// ---------------------------------------------------------------------------
// Softmax over rows of 40, one wave per row
// ---------------------------------------------------------------------------
__global__ void k_softmax40(const float* __restrict__ logits, float* __restrict__ probs, int Nr)
{
    int t = blockIdx.x * blockDim.x + threadIdx.x;
    int row = t >> 5, lane = t & 31;
    if (row >= Nr) return;
    const float* in = logits + (size_t)row * 40;
    float v0 = in[lane];
    float v1 = (lane + 32 < 40) ? in[lane + 32] : -INFINITY;
    float m = fmaxf(v0, v1);
    #pragma unroll
    for (int off = 16; off > 0; off >>= 1) m = fmaxf(m, __shfl_xor(m, off));
    float e0 = __expf(v0 - m);
    float e1 = (lane + 32 < 40) ? __expf(v1 - m) : 0.0f;
    float s = e0 + e1;
    #pragma unroll
    for (int off = 16; off > 0; off >>= 1) s += __shfl_xor(s, off);
    float inv = 1.0f / s;
    float* o = probs + (size_t)row * 40;
    o[lane] = e0 * inv;
    if (lane + 32 < 40) o[lane + 32] = e1 * inv;
}

// ---------------------------------------------------------------------------
extern "C" void kernel_launch(void* const* d_in, const int* in_sizes, int n_in,
                              void* d_out, int out_size, void* d_ws, size_t ws_size,
                              hipStream_t stream)
{
    const float* x   = (const float*)d_in[0];
    const int*   ei  = (const int*)d_in[1];
    const float* Wg1 = (const float*)d_in[2];  const float* bg1 = (const float*)d_in[3];
    const float* Wg2 = (const float*)d_in[4];  const float* bg2 = (const float*)d_in[5];
    const float* Wl1 = (const float*)d_in[6];  const float* bl1 = (const float*)d_in[7];
    const float* Wl2 = (const float*)d_in[8];  const float* bl2 = (const float*)d_in[9];
    const float* Wf  = (const float*)d_in[10]; const float* bf  = (const float*)d_in[11];
    float* out = (float*)d_out;

    const int DIM_IN = 256, H = 128, DIM_OUT = 40;
    const int Nn = in_sizes[0] / DIM_IN;     // 50000
    const int E  = in_sizes[1] / 2;          // 800000
    const int* srcp = ei;
    const int* dstp = ei + E;

    // workspace carve-up (floats): dinv[N] | bufA[N*128] | bufB[N*128] | hcat[N*256]
    float* dinv = (float*)d_ws;
    float* bufA = dinv + Nn;
    float* bufB = bufA + (size_t)Nn * H;
    float* hcat = bufB + (size_t)Nn * H;

    const int TB = 256;
    dim3 blk(TB);
    int gN     = (Nn + TB - 1) / TB;
    int gE     = (E + TB - 1) / TB;
    int gNF    = (int)(((size_t)Nn * H + TB - 1) / TB);
    int gEdge  = (int)(((size_t)E * 32 + TB - 1) / TB);
    int gGemm  = (Nn + 63) / 64;
    int gSmax  = (int)(((size_t)Nn * 32 + TB - 1) / TB);

    // 1) normalization: dinv = rsqrt(1 + in_degree)
    k_fill_ones<<<gN, blk, 0, stream>>>(dinv, Nn);
    k_deg_accum<<<gE, blk, 0, stream>>>(dstp, E, dinv);
    k_rsqrt<<<gN, blk, 0, stream>>>(dinv, Nn);

    // 2) GCN layer 1: bufA = x @ Wg1
    k_gemm_wmma<128><<<gGemm, dim3(256), 0, stream>>>(
        x, DIM_IN, Wg1, DIM_IN, H, nullptr, bufA, H, 0, Nn, 0);
    // self-loop + scatter + bias/relu -> h1 in bufB
    k_init_self<<<gNF, blk, 0, stream>>>(dinv, bufA, bufB, H, nullptr, Nn);
    k_scatter_edges<<<gEdge, blk, 0, stream>>>(srcp, dstp, E, dinv, bufA, bufB, H);
    k_bias_relu<<<gNF, blk, 0, stream>>>(bufB, bg1, Nn);

    // 3) GCN layer 2: bufA = h1 @ Wg2 ; result (+bg2) into hcat[:, 0:128]
    k_gemm_wmma<128><<<gGemm, dim3(256), 0, stream>>>(
        bufB, H, Wg2, H, H, nullptr, bufA, H, 0, Nn, 0);
    k_init_self<<<gNF, blk, 0, stream>>>(dinv, bufA, hcat, 2 * H, bg2, Nn);
    k_scatter_edges<<<gEdge, blk, 0, stream>>>(srcp, dstp, E, dinv, bufA, hcat, 2 * H);

    // 4) Linear branch: bufB = relu(x @ Wl1 + bl1); hcat[:,128:256] = bufB @ Wl2 + bl2
    k_gemm_wmma<128><<<gGemm, dim3(256), 0, stream>>>(
        x, DIM_IN, Wl1, DIM_IN, H, bl1, bufB, H, 0, Nn, 1 /*relu epi*/);
    k_gemm_wmma<128><<<gGemm, dim3(256), 0, stream>>>(
        bufB, H, Wl2, H, H, bl2, hcat, 2 * H, H, Nn, 0);

    // 5) Head: logits = relu(hcat) @ Wf + bf   (relu fused into A load)
    k_gemm_wmma<64><<<gGemm, dim3(128), 0, stream>>>(
        hcat, 2 * H, Wf, 2 * H, DIM_OUT, bf, out, DIM_OUT, 0, Nn, 2 /*relu on load*/);

    // 6) softmax -> second output
    k_softmax40<<<gSmax, blk, 0, stream>>>(out, out + (size_t)Nn * DIM_OUT, Nn);
}
